// MemoryNet_10436770530030
// MI455X (gfx1250) — matrix operand order repaired and checked by reading, not compiled, not run on hardware
//
#include <hip/hip_runtime.h>
#include <stdint.h>

// ---------------------------------------------------------------------------
// LIF/LI spiking RNN for MI455X (gfx1250, wave32, WMMA).
//
// Phase 0: convert x and weights to bf16 in workspace (memory-bound passes;
//          everything L2-resident afterwards: 192 MB L2 >> ~100 MB working set)
// Phase 1: cur[t,b,h] = x[b,t,:] @ w_in^T   -- one big bf16 WMMA GEMM,
//          A and B fragments are pure global_load_b128 of bf16 (no VALU cvt
//          in the K-loop).
// Phase 2: persistent scan kernel, 8 WGs (one 64-wide hidden slice each),
//          w_rec slice resident in LDS (64 KB bf16), per-step device barrier
//          via atomic counters + s_sleep backoff, z exchanged through a
//          triple-buffered bf16 buffer in L2. The recurrent WMMA (which only
//          needs z_{t-1}) is overlapped with the barrier wait.
// ---------------------------------------------------------------------------

typedef __attribute__((ext_vector_type(16))) __bf16 v16bf;
typedef __attribute__((ext_vector_type(8)))  float  v8f;

struct __align__(16) U4 { unsigned int x, y, z, w; };
struct __align__(16) F4 { float x, y, z, w; };

union FragBF {
  v16bf v;
  U4 q[2];
  unsigned short u[16];
};

#define B_   64
#define T_   512
#define H_   512
#define O_   128
#define NWG  8
#define HS   (H_ / NWG)   // 64 hidden cols per WG
#define OS   (O_ / NWG)   // 16 output cols per WG

#define DT_          0.01f
#define TAU_SYN_INV_ 200.0f
#define TAU_MEM_INV_ 100.0f
#define V_LEAK_      0.0f
#define V_TH_        1.0f
#define V_RESET_     0.0f
#define DTM_   (DT_ * TAU_MEM_INV_)          // 1.0
#define DECI_  (1.0f - DT_ * TAU_SYN_INV_)   // -1.0

__device__ __forceinline__ unsigned short f2bf(float f) {
  unsigned int u = __float_as_uint(f);
  u += 0x7fffu + ((u >> 16) & 1u);
  return (unsigned short)(u >> 16);
}

__device__ __forceinline__ v8f wmma_bf16(const FragBF& a, const FragBF& b, v8f c) {
  return __builtin_amdgcn_wmma_f32_16x16x32_bf16(
      /*neg_a=*/false, a.v, /*neg_b=*/false, b.v,
      /*c_mod=*/(short)0, c, /*reuse_a=*/false, /*reuse_b=*/false);
}

// ---------------------------------------------------------------------------
__global__ void k_init(int* cnt) {
  int i = blockIdx.x * blockDim.x + threadIdx.x;
  if (i < T_) cnt[i] = 0;
}

// fp32 -> bf16 weight conversion (w_in | w_rec | w_out packed into ws)
__global__ void k_cvt(const float* __restrict__ w_in,
                      const float* __restrict__ w_rec,
                      const float* __restrict__ w_out,
                      unsigned short* __restrict__ bf) {
  int i = blockIdx.x * blockDim.x + threadIdx.x;
  const int nin = H_ * H_;
  if (i < nin) {
    bf[i]        = f2bf(w_in[i]);
    bf[nin + i]  = f2bf(w_rec[i]);
    if (i < O_ * H_) bf[2 * nin + i] = f2bf(w_out[i]);
  }
}

// fp32 -> bf16 conversion of x, 8 elements / thread, vectorized 16B ld/st
__global__ __launch_bounds__(256) void k_cvtx(const float* __restrict__ x,
                                              unsigned short* __restrict__ xbf) {
  const size_t i = ((size_t)blockIdx.x * blockDim.x + threadIdx.x) * 8;
  const F4* src = (const F4*)(x + i);
  F4 f0 = src[0], f1 = src[1];
  U4 o;
  o.x = (unsigned)f2bf(f0.x) | ((unsigned)f2bf(f0.y) << 16);
  o.y = (unsigned)f2bf(f0.z) | ((unsigned)f2bf(f0.w) << 16);
  o.z = (unsigned)f2bf(f1.x) | ((unsigned)f2bf(f1.y) << 16);
  o.w = (unsigned)f2bf(f1.z) | ((unsigned)f2bf(f1.w) << 16);
  *(U4*)(xbf + i) = o;
}

// ---------------------------------------------------------------------------
// Phase 1: cur[t][b][h] = sum_k x[b][t][k] * w_in[h][k]
// grid = (H/64, T), block = 256 (8 waves). Each WG: 64 batches x 64 h cols.
// Wave w -> M-tile mt = w&3, N-tiles {(w>>2)*2, (w>>2)*2+1}.
__global__ __launch_bounds__(256) void k_ingemm(const unsigned short* __restrict__ xbf,
                                                const unsigned short* __restrict__ win_bf,
                                                float* __restrict__ cur) {
  const int t   = blockIdx.y;
  const int hb  = blockIdx.x;             // 64-col slice
  const int tid = threadIdx.x;
  const int wv = tid >> 5, lane = tid & 31;
  const int ml = lane & 15, kh = lane >> 4;
  const int mt = wv & 3;
  const int nt0 = (wv >> 2) * 2;

  v8f acc0 = {}; v8f acc1 = {};
  const int b_row = mt * 16 + ml;
  const unsigned short* xrow = xbf + ((size_t)b_row * T_ + t) * H_;
  const unsigned short* b0row = win_bf + (size_t)(hb * 64 + nt0 * 16 + ml) * H_;
  const unsigned short* b1row = win_bf + (size_t)(hb * 64 + (nt0 + 1) * 16 + ml) * H_;

  for (int kc = 0; kc < H_ / 32; ++kc) {
    const int k0 = kc * 32 + kh * 16;     // A lanes 0-15: K 0-7 ; lanes 16-31: K 16-23
    const int k1 = k0 + 8;                // and K 8-15 / 24-31
    FragBF a;
    a.q[0] = *(const U4*)(xrow + k0);
    a.q[1] = *(const U4*)(xrow + k1);
    const int kb = kc * 32 + kh * 16;     // B lanes 0-15: K 0-15 ; lanes 16-31: K 16-31
    FragBF b0, b1;
    b0.q[0] = *(const U4*)(b0row + kb); b0.q[1] = *(const U4*)(b0row + kb + 8);
    b1.q[0] = *(const U4*)(b1row + kb); b1.q[1] = *(const U4*)(b1row + kb + 8);
    acc0 = wmma_bf16(a, b0, acc0);
    acc1 = wmma_bf16(a, b1, acc1);
  }

  float* out = cur + (size_t)t * B_ * H_;
#pragma unroll
  for (int e = 0; e < 8; ++e) {
    const int bo = mt * 16 + kh * 8 + e;  // C: VGPR e -> M = e (+8 for hi lanes)
    out[(size_t)bo * H_ + hb * 64 + nt0 * 16 + ml]       = acc0[e];
    out[(size_t)bo * H_ + hb * 64 + (nt0 + 1) * 16 + ml] = acc1[e];
  }
}

// ---------------------------------------------------------------------------
// Phase 2: sequential scan. 8 persistent WGs x 256 threads.
__global__ __launch_bounds__(256) void k_scan(const float* __restrict__ cur,
                                              const unsigned short* __restrict__ wrec_bf,
                                              const unsigned short* __restrict__ wout_bf,
                                              unsigned short* __restrict__ zbuf,
                                              int* __restrict__ cnt,
                                              float* __restrict__ readouts,
                                              float* __restrict__ spikes) {
  __shared__ __align__(16) unsigned short lds_wrec[HS * H_];   // 64 KB bf16 w_rec slice

  const int hb  = blockIdx.x;
  const int tid = threadIdx.x;
  const int wv = tid >> 5, lane = tid & 31;
  const int ml = lane & 15, kh = lane >> 4;
  const int mt = wv & 3;
  const int nt0 = (wv >> 2) * 2;

  // Stage w_rec slice into LDS (rows = local hidden col, contiguous K)
  {
    const U4* src = (const U4*)(wrec_bf + (size_t)hb * HS * H_);
    U4* dst = (U4*)lds_wrec;
    for (int i = tid; i < HS * H_ / 8; i += 256) dst[i] = src[i];
  }
  __syncthreads();

  // Recurrent state in WMMA C layout: tiles (mt, nt0) and (mt, nt0+1)
  v8f v0 = {}, v1 = {}, i0 = {}, i1 = {};
  // Readout state (waves 0-3 only; mt == wv)
  v8f vr = {}, ir = {};

  const int b_base = mt * 16 + kh * 8;
  const int h0 = hb * 64 + nt0 * 16 + ml;
  const int h1 = h0 + 16;
  const unsigned short* worow = wout_bf + (size_t)(hb * OS + ml) * H_;

#pragma unroll 1
  for (int t = 0; t < T_; ++t) {
    // --- 1. membrane decay + spike + reset (elementwise, fragment layout) ---
    float z0[8], z1[8];
#pragma unroll
    for (int e = 0; e < 8; ++e) {
      float vd = v0[e] + DTM_ * ((V_LEAK_ - v0[e]) + i0[e]);
      float z  = (vd - V_TH_ > 0.0f) ? 1.0f : 0.0f;
      z0[e] = z; v0[e] = (1.0f - z) * vd + z * V_RESET_;
      vd = v1[e] + DTM_ * ((V_LEAK_ - v1[e]) + i1[e]);
      z  = (vd - V_TH_ > 0.0f) ? 1.0f : 0.0f;
      z1[e] = z; v1[e] = (1.0f - z) * vd + z * V_RESET_;
    }

    // --- 2. publish z: fp32 spikes output + bf16 exchange buffer ---
    unsigned short* zb = zbuf + (size_t)(t % 3) * B_ * H_;
    float* sp = spikes + (size_t)t * B_ * H_;
#pragma unroll
    for (int e = 0; e < 8; ++e) {
      const int b = b_base + e;
      sp[(size_t)b * H_ + h0] = z0[e];
      sp[(size_t)b * H_ + h1] = z1[e];
      zb[(size_t)b * H_ + h0] = f2bf(z0[e]);
      zb[(size_t)b * H_ + h1] = f2bf(z1[e]);
    }

    // --- 3. signal arrival (release) ---
    __threadfence();
    __syncthreads();
    if (tid == 0) atomicAdd(&cnt[t], 1);

    // --- 4. i_syn decay + input current + recurrent GEMM (uses z_{t-1}) ---
    const float* cr = cur + (size_t)t * B_ * H_;
#pragma unroll
    for (int e = 0; e < 8; ++e) {
      const int b = b_base + e;
      i0[e] = i0[e] * DECI_ + cr[(size_t)b * H_ + h0];
      i1[e] = i1[e] * DECI_ + cr[(size_t)b * H_ + h1];
    }
    if (t > 0) {
      const unsigned short* arow =
          zbuf + (size_t)((t + 2) % 3) * B_ * H_ + (size_t)(mt * 16 + ml) * H_;
      for (int kc = 0; kc < H_ / 32; ++kc) {
        const int k0 = kc * 32 + kh * 16, k1 = k0 + 8;
        const int kb = kc * 32 + kh * 16;
        FragBF a;
        a.q[0] = *(const U4*)(arow + k0);
        a.q[1] = *(const U4*)(arow + k1);
        FragBF b0, b1;
        b0.q[0] = *(const U4*)&lds_wrec[(nt0 * 16 + ml) * H_ + kb];
        b0.q[1] = *(const U4*)&lds_wrec[(nt0 * 16 + ml) * H_ + kb + 8];
        b1.q[0] = *(const U4*)&lds_wrec[((nt0 + 1) * 16 + ml) * H_ + kb];
        b1.q[1] = *(const U4*)&lds_wrec[((nt0 + 1) * 16 + ml) * H_ + kb + 8];
        i0 = wmma_bf16(a, b0, i0);
        i1 = wmma_bf16(a, b1, i1);
      }
    }
    if (t + 1 < T_)
      __builtin_prefetch(cur + (size_t)(t + 1) * B_ * H_ + (size_t)b_base * H_ + h0, 0, 0);

    // --- 5. wait for all WGs' z of this step (acquire) ---
    if (tid == 0) {
      while (__hip_atomic_load(&cnt[t], __ATOMIC_ACQUIRE, __HIP_MEMORY_SCOPE_AGENT) < NWG)
        __builtin_amdgcn_s_sleep(1);
    }
    __syncthreads();
    __threadfence();

    // --- 6. readout LI cell (waves 0-3): v_r uses OLD i_r; i_r += z_t @ w_out^T ---
    if (wv < 4) {
      float* ro = readouts + (size_t)t * B_ * O_;
#pragma unroll
      for (int e = 0; e < 8; ++e) {
        const float vrn = vr[e] + DTM_ * ((V_LEAK_ - vr[e]) + ir[e]);
        ro[(size_t)(b_base + e) * O_ + hb * OS + ml] = vrn;
        vr[e] = vrn;
      }
      v8f c;
#pragma unroll
      for (int e = 0; e < 8; ++e) c[e] = ir[e] * DECI_;
      const unsigned short* arow =
          zbuf + (size_t)(t % 3) * B_ * H_ + (size_t)(mt * 16 + ml) * H_;
      for (int kc = 0; kc < H_ / 32; ++kc) {
        const int k0 = kc * 32 + kh * 16, k1 = k0 + 8;
        const int kb = kc * 32 + kh * 16;
        FragBF a, bo;
        a.q[0]  = *(const U4*)(arow + k0);
        a.q[1]  = *(const U4*)(arow + k1);
        bo.q[0] = *(const U4*)(worow + kb);
        bo.q[1] = *(const U4*)(worow + kb + 8);
        c = wmma_bf16(a, bo, c);
      }
      ir = c;
    }
  }
}

// ---------------------------------------------------------------------------
extern "C" void kernel_launch(void* const* d_in, const int* in_sizes, int n_in,
                              void* d_out, int out_size, void* d_ws, size_t ws_size,
                              hipStream_t stream) {
  const float* x     = (const float*)d_in[0];
  const float* w_in  = (const float*)d_in[1];
  const float* w_rec = (const float*)d_in[2];
  const float* w_out = (const float*)d_in[3];
  (void)in_sizes; (void)n_in; (void)out_size; (void)ws_size;

  // Workspace layout:
  //   cur   : T*B*H fp32                                (64 MB)
  //   xbf   : B*T*H bf16                                (32 MB)
  //   bf    : w_in | w_rec | w_out bf16                 (~1.1 MB)
  //   zbuf  : 3*B*H bf16                                (192 KB)
  //   cnt   : T int                                     (2 KB)
  char* ws = (char*)d_ws;
  float* cur = (float*)ws;
  unsigned short* xbf = (unsigned short*)(ws + (size_t)T_ * B_ * H_ * 4);
  unsigned short* bf  = xbf + (size_t)B_ * T_ * H_;
  unsigned short* win_bf  = bf;
  unsigned short* wrec_bf = bf + (size_t)H_ * H_;
  unsigned short* wout_bf = bf + (size_t)2 * H_ * H_;
  unsigned short* zbuf    = bf + (size_t)2 * H_ * H_ + O_ * H_;
  int* cnt = (int*)(bf + (size_t)2 * H_ * H_ + O_ * H_ + 3 * B_ * H_);

  float* readouts = (float*)d_out;                        // [T,B,O]
  float* spikes   = (float*)d_out + (size_t)T_ * B_ * O_; // [T,B,H]

  k_init<<<2, 256, 0, stream>>>(cnt);
  k_cvt<<<(H_ * H_ + 255) / 256, 256, 0, stream>>>(w_in, w_rec, w_out, bf);
  k_cvtx<<<((size_t)B_ * T_ * H_ / 8 + 255) / 256, 256, 0, stream>>>(x, xbf);
  k_ingemm<<<dim3(H_ / 64, T_), 256, 0, stream>>>(xbf, win_bf, cur);
  k_scan<<<NWG, 256, 0, stream>>>(cur, wrec_bf, wout_bf, zbuf, cnt, readouts, spikes);
}
